// PointNetSetAbstraction_72902774882330
// MI455X (gfx1250) — matrix operand order repaired and checked by reading, not compiled, not run on hardware
//
#include <hip/hip_runtime.h>
#include <math.h>

// ---------------- problem constants ----------------
#define BB 4
#define NN 8192
#define DD 64
#define SS 2048            // NPOINT
#define NS 32              // NSAMPLE
#define KIN 128            // 2*D
#define MROWS (BB*SS*NS)   // 262144 rows into the MLP
#define R2 0.01f           // RADIUS^2
#define BN_EPS 1e-5f
#define NORM_EPS 1e-6f
#define NSTAT1 4194304.0f  // S*NS*D elements per batch for std1/std2

typedef __attribute__((ext_vector_type(16))) _Float16 v16h;
typedef __attribute__((ext_vector_type(8)))  float    v8f;
typedef __attribute__((ext_vector_type(4)))  float    f4_t;

union Frag16 { v16h v; f4_t f[2]; };

// ---------------- 0: zero the accumulators ----------------
__global__ void k_zero(float* acc) {
    int i = blockIdx.x * 256 + threadIdx.x;
    if (i < 2048) acc[i] = 0.0f;
}

// ---------------- 1: farthest point sampling (1 block per batch) ---------
__global__ __launch_bounds__(1024)
void k_fps(const float* __restrict__ xyz, int* __restrict__ fps_idx) {
    const int b = blockIdx.x;
    const float* base = xyz + (size_t)b * 3 * NN;
    const int tid = threadIdx.x;
    float px[8], py[8], pz[8], dist[8];
#pragma unroll
    for (int k = 0; k < 8; ++k) {
        int j = tid + k * 1024;
        px[k] = base[j]; py[k] = base[NN + j]; pz[k] = base[2 * NN + j];
        dist[k] = 1e10f;
    }
    __shared__ float swv[32];
    __shared__ int   swi[32];
    __shared__ int   sfar;
    int far = 0;
    for (int i = 0; i < SS; ++i) {
        if (tid == 0) fps_idx[b * SS + i] = far;
        float cx = base[far], cy = base[NN + far], cz = base[2 * NN + far];
        float bestv = -1.0f; int besti = 0;
#pragma unroll
        for (int k = 0; k < 8; ++k) {
            float dx = px[k] - cx, dy = py[k] - cy, dz = pz[k] - cz;
            float d = dx * dx + dy * dy + dz * dz;
            dist[k] = fminf(dist[k], d);
            if (dist[k] > bestv) { bestv = dist[k]; besti = tid + k * 1024; }
        }
        for (int off = 16; off > 0; off >>= 1) {
            float ov = __shfl_down(bestv, off, 32);
            int   oi = __shfl_down(besti, off, 32);
            if (ov > bestv) { bestv = ov; besti = oi; }
        }
        if ((tid & 31) == 0) { swv[tid >> 5] = bestv; swi[tid >> 5] = besti; }
        __syncthreads();
        if (tid < 32) {
            bestv = swv[tid]; besti = swi[tid];
            for (int off = 16; off > 0; off >>= 1) {
                float ov = __shfl_down(bestv, off, 32);
                int   oi = __shfl_down(besti, off, 32);
                if (ov > bestv) { bestv = ov; besti = oi; }
            }
            if (tid == 0) sfar = besti;
        }
        __syncthreads();
        far = sfar;
    }
}

// ---------------- 2: gather new_xyz / new_points, mean1, write xyz^T out --
__global__ void k_gather(const float* __restrict__ xyz, const float* __restrict__ pts,
                         const int* __restrict__ fps_idx,
                         float* __restrict__ new_xyz, float* __restrict__ new_pts,
                         float* __restrict__ mean1, float* __restrict__ out) {
    int X = blockIdx.x * 256 + threadIdx.x;      // b*S + s
    if (X >= BB * SS) return;
    int b = X >> 11, s = X & (SS - 1);
    int idx = fps_idx[X];
    for (int d = 0; d < 3; ++d) {
        float v = xyz[(size_t)b * 3 * NN + (size_t)d * NN + idx];
        new_xyz[X * 3 + d] = v;
        out[(size_t)b * 3 * SS + (size_t)d * SS + s] = v;   // new_xyz^T (B,3,S)
    }
    float acc = 0.0f;
    for (int c = 0; c < DD; ++c) {
        float v = pts[(size_t)b * DD * NN + (size_t)c * NN + idx];
        new_pts[(size_t)X * DD + c] = v;
        acc += v;
    }
    mean1[X] = acc * (1.0f / DD);
}

// ---------------- 3: ball query (1 wave per center; ordered scan) --------
__global__ __launch_bounds__(256)
void k_query(const float* __restrict__ xyz, const float* __restrict__ new_xyz,
             int* __restrict__ gidx) {
    int w = threadIdx.x >> 5;
    int lane = threadIdx.x & 31;
    int X = blockIdx.x * 8 + w;                  // b*S + s
    int b = X >> 11;
    float cx = new_xyz[X * 3 + 0], cy = new_xyz[X * 3 + 1], cz = new_xyz[X * 3 + 2];
    const float* base = xyz + (size_t)b * 3 * NN;
    __shared__ int slots[8][32];
    int total = 0;
    for (int j0 = 0; j0 < NN; j0 += 32) {
        int j = j0 + lane;
        float dx = base[j] - cx, dy = base[NN + j] - cy, dz = base[2 * NN + j] - cz;
        bool in = (dx * dx + dy * dy + dz * dz) <= R2;
        unsigned mask = (unsigned)__ballot(in);
        if (in) {
            int pos = total + __popc(mask & ((1u << lane) - 1u));
            if (pos < NS) slots[w][pos] = j;
        }
        total += __popc(mask);
        if (total >= NS) break;
    }
    __syncthreads();
    int v;
    if (total == 0) v = NN - 1;
    else { int first = slots[w][0]; v = (lane < total) ? slots[w][lane] : first; }
    gidx[(size_t)X * NS + lane] = v;
}

__device__ inline float std_unb(const float* acc, int off, int b) {
    float s = acc[off + b], q = acc[off + 4 + b];
    float var = (q - s * s / NSTAT1) / (NSTAT1 - 1.0f);
    return sqrtf(fmaxf(var, 0.0f));
}

// ---------------- 4: per-batch sum/sumsq of (grouped - mean1) ------------
__global__ void k_stats1(const float* __restrict__ pts, const int* __restrict__ gidx,
                         const float* __restrict__ mean1, float* __restrict__ acc) {
    size_t idx = (size_t)blockIdx.x * 256 + threadIdx.x;
    const size_t stride = 512 * 256;             // divides per-batch extent
    int b = (int)(idx >> 22);
    float s = 0.0f, q = 0.0f;
    for (size_t i = idx; i < (size_t)BB * SS * NS * DD; i += stride) {
        int c = (int)(i & 63);
        float v = pts[(size_t)b * DD * NN + (size_t)c * NN + gidx[i >> 6]] - mean1[i >> 11];
        s += v; q += v * v;
    }
    atomicAdd(&acc[b], s);
    atomicAdd(&acc[4 + b], q);
}

// ---------------- 5: mean2 over nsample of gp ----------------------------
__global__ void k_mean2(const float* __restrict__ pts, const int* __restrict__ gidx,
                        const float* __restrict__ mean1, const float* __restrict__ acc,
                        const float* __restrict__ a1, const float* __restrict__ be1,
                        float* __restrict__ mean2) {
    int g = blockIdx.x * 256 + threadIdx.x;      // (b*S+s)*64 + c
    int c = g & 63, X = g >> 6, b = X >> 11;
    float inv1 = 1.0f / (std_unb(acc, 0, b) + NORM_EPS);
    float m1 = mean1[X];
    const int* gi = gidx + (size_t)X * NS;
    float al = a1[c], bt = be1[c];
    float s = 0.0f;
    for (int n = 0; n < NS; ++n) {
        float gr = pts[(size_t)b * DD * NN + (size_t)c * NN + gi[n]];
        s += al * (gr - m1) * inv1 + bt;
    }
    mean2[g] = s * (1.0f / NS);
}

// ---------------- 6: per-batch sum/sumsq of (gp - mean2) -----------------
__global__ void k_stats2(const float* __restrict__ pts, const int* __restrict__ gidx,
                         const float* __restrict__ mean1, const float* __restrict__ mean2,
                         const float* __restrict__ a1, const float* __restrict__ be1,
                         float* __restrict__ acc) {
    size_t idx = (size_t)blockIdx.x * 256 + threadIdx.x;
    const size_t stride = 512 * 256;
    int b = (int)(idx >> 22);
    float inv1 = 1.0f / (std_unb(acc, 0, b) + NORM_EPS);
    float s = 0.0f, q = 0.0f;
    for (size_t i = idx; i < (size_t)BB * SS * NS * DD; i += stride) {
        int c = (int)(i & 63);
        float gr = pts[(size_t)b * DD * NN + (size_t)c * NN + gidx[i >> 6]];
        float gp = a1[c] * (gr - mean1[i >> 11]) * inv1 + be1[c];
        float v = gp - mean2[(i >> 11) * 64 + c];
        s += v; q += v * v;
    }
    atomicAdd(&acc[8 + b], s);
    atomicAdd(&acc[12 + b], q);
}

// ---------------- 7: build concat input x0 (f16) -------------------------
__global__ void k_buildx0(const float* __restrict__ pts, const int* __restrict__ gidx,
                          const float* __restrict__ mean1, const float* __restrict__ mean2,
                          const float* __restrict__ new_pts, const float* __restrict__ acc,
                          const float* __restrict__ a1, const float* __restrict__ be1,
                          const float* __restrict__ a2, const float* __restrict__ be2,
                          _Float16* __restrict__ x0) {
    int r = blockIdx.x * 256 + threadIdx.x;      // row = (b*S+s)*NS + n
    int X = r >> 5, b = X >> 11;
    float inv1 = 1.0f / (std_unb(acc, 0, b) + NORM_EPS);
    float inv2 = 1.0f / (std_unb(acc, 8, b) + NORM_EPS);
    int gi = gidx[r];
    float m1 = mean1[X];
    _Float16* out = x0 + (size_t)r * KIN;
    for (int c = 0; c < DD; ++c) {
        float gr = pts[(size_t)b * DD * NN + (size_t)c * NN + gi];
        float gp = a1[c] * (gr - m1) * inv1 + be1[c];
        out[c] = (_Float16)gp;
        float np = new_pts[(size_t)X * DD + c];
        float m2 = mean2[(size_t)X * DD + c];
        float npn = a2[c] * ((np - m2) * inv2) + be2[c];
        out[DD + c] = (_Float16)npn;
    }
}

// ---------------- 8: weight f32 -> f16 -----------------------------------
__global__ void k_cvtw(const float* __restrict__ w, _Float16* __restrict__ wh, int count) {
    int i = blockIdx.x * 256 + threadIdx.x;
    if (i < count) wh[i] = (_Float16)w[i];
}

// ---------------- 9: WMMA GEMM  Y[M,O] = X[M,128] * W[O,128]^T + bias ----
__global__ __launch_bounds__(256)
void k_gemm(const _Float16* __restrict__ X, const _Float16* __restrict__ W,
            const float* __restrict__ bias, float* __restrict__ Y, int O) {
    const int wave = threadIdx.x >> 5;
    const int lane = threadIdx.x & 31;
    const int hi = lane >> 4;
    const int m = lane & 15;
    const int mBase = blockIdx.y * 128 + wave * 16;
    const int oBase = blockIdx.x * 16;
    const _Float16* xrow = X + (size_t)(mBase + m) * KIN;
    const _Float16* wrow = W + (size_t)(oBase + m) * KIN;   // column (out channel)
    v8f acc = {};
#pragma unroll
    for (int kk = 0; kk < 4; ++kk) {
        const int kb = kk * 32;
        Frag16 a, bfr;
        // A 16x32 f16: lanes 0-15 hold K {0..7,16..23}, lanes 16-31 {8..15,24..31}
        a.f[0] = *(const f4_t*)(xrow + kb + hi * 8);
        a.f[1] = *(const f4_t*)(xrow + kb + 16 + hi * 8);
        // B 32x16 f16: same K split as A: lanes 0-15 K {0..7,16..23}, lanes 16-31 {8..15,24..31}
        bfr.f[0] = *(const f4_t*)(wrow + kb + hi * 8);
        bfr.f[1] = *(const f4_t*)(wrow + kb + 16 + hi * 8);
        acc = __builtin_amdgcn_wmma_f32_16x16x32_f16(
            false, a.v, false, bfr.v, (short)0, acc, false, false);
    }
    float bv = bias[oBase + m];
#pragma unroll
    for (int v = 0; v < 8; ++v) {
        int row = mBase + v + hi * 8;
        Y[(size_t)row * O + (oBase + m)] = acc[v] + bv;
    }
}

// ---------------- 10: per-channel sum / sumsq for BN ---------------------
__global__ void k_chanstats(const float* __restrict__ Y, float* __restrict__ acc,
                            int O, size_t total) {
    size_t idx = (size_t)blockIdx.x * 256 + threadIdx.x;
    const size_t stride = 512 * 256;             // divisible by 128 and 256
    int c = (int)(idx % O);
    float s = 0.0f, q = 0.0f;
    for (size_t i = idx; i < total; i += stride) {
        float v = Y[i]; s += v; q += v * v;
    }
    atomicAdd(&acc[c], s);
    atomicAdd(&acc[256 + c], q);
}

// ---------------- 11: BN affine + ReLU + cvt f16 (layers 0,1; O=128) -----
__global__ void k_bnrelu(const float* __restrict__ Y, const float* __restrict__ acc,
                         const float* __restrict__ g, const float* __restrict__ bb,
                         _Float16* __restrict__ Xo) {
    const int O = 128;
    int tid = threadIdx.x;
    int c = tid & (O - 1);
    int sub = tid >> 7;                          // 0..1
    float mean = acc[c] * (1.0f / MROWS);
    float var = acc[256 + c] * (1.0f / MROWS) - mean * mean;
    float aS = g[c] * rsqrtf(var + BN_EPS);
    float bS = bb[c] - mean * aS;
    size_t r0 = (size_t)blockIdx.x * 32;
    for (int r = sub; r < 32; r += 2) {
        size_t i = (r0 + r) * O + c;
        float v = aS * Y[i] + bS;
        Xo[i] = (_Float16)(v > 0.0f ? v : 0.0f);
    }
}

// ---------------- 12: final BN + ReLU + max over nsample, transpose out --
__global__ void k_final(const float* __restrict__ Y, const float* __restrict__ acc,
                        const float* __restrict__ g, const float* __restrict__ bb,
                        float* __restrict__ out) {
    const int O = 256;
    int o = threadIdx.x;
    int b = blockIdx.x >> 11, s = blockIdx.x & (SS - 1);
    float mean = acc[o] * (1.0f / MROWS);
    float var = acc[256 + o] * (1.0f / MROWS) - mean * mean;
    float aS = g[o] * rsqrtf(var + BN_EPS);
    float bS = bb[o] - mean * aS;
    size_t rowBase = (size_t)blockIdx.x * NS;
    float mx = 0.0f;                             // relu >= 0
    for (int n = 0; n < NS; ++n) {
        float v = aS * Y[(rowBase + n) * O + o] + bS;
        mx = fmaxf(mx, v);
    }
    out[(size_t)BB * 3 * SS + ((size_t)b * O + o) * SS + s] = mx;
}

// ---------------- host launcher ------------------------------------------
extern "C" void kernel_launch(void* const* d_in, const int* in_sizes, int n_in,
                              void* d_out, int out_size, void* d_ws, size_t ws_size,
                              hipStream_t stream) {
    (void)in_sizes; (void)n_in; (void)out_size; (void)ws_size;
    const float* xyz = (const float*)d_in[0];
    const float* pts = (const float*)d_in[1];
    const float* w[3]  = {(const float*)d_in[2], (const float*)d_in[6],  (const float*)d_in[10]};
    const float* bi[3] = {(const float*)d_in[3], (const float*)d_in[7],  (const float*)d_in[11]};
    const float* gg[3] = {(const float*)d_in[4], (const float*)d_in[8],  (const float*)d_in[12]};
    const float* bbn[3]= {(const float*)d_in[5], (const float*)d_in[9],  (const float*)d_in[13]};
    const float* a1 = (const float*)d_in[14];
    const float* be1 = (const float*)d_in[15];
    const float* a2 = (const float*)d_in[16];
    const float* be2 = (const float*)d_in[17];
    float* out = (float*)d_out;

    char* ws = (char*)d_ws;
    int*   fps_idx = (int*)(ws + 0);                         //  32 KB
    int*   gidx    = (int*)(ws + 32768);                     //   1 MB
    float* new_xyz = (float*)(ws + 1081344);                 //  96 KB
    float* new_pts = (float*)(ws + 1179648);                 //   2 MB
    float* mean1   = (float*)(ws + 3276800);                 //  32 KB
    float* mean2   = (float*)(ws + 3309568);                 //   2 MB
    float* accums  = (float*)(ws + 5406720);                 //   8 KB (batch + channel stats)
    _Float16* wf16 = (_Float16*)(ws + 5414912);              //  64 KB
    _Float16* xbuf = (_Float16*)(ws + 5480448);              //  67 MB (reused in place)
    float* ybuf    = (float*)(ws + 72589312);                // 268 MB

    float* chan0 = accums + 16;
    float* chan1 = accums + 16 + 512;
    float* chan2 = accums + 16 + 1024;

    k_zero<<<8, 256, 0, stream>>>(accums);
    k_fps<<<BB, 1024, 0, stream>>>(xyz, fps_idx);
    k_gather<<<(BB * SS) / 256, 256, 0, stream>>>(xyz, pts, fps_idx, new_xyz, new_pts, mean1, out);
    k_query<<<(BB * SS) / 8, 256, 0, stream>>>(xyz, new_xyz, gidx);
    k_stats1<<<512, 256, 0, stream>>>(pts, gidx, mean1, accums);
    k_mean2<<<(BB * SS * DD) / 256, 256, 0, stream>>>(pts, gidx, mean1, accums, a1, be1, mean2);
    k_stats2<<<512, 256, 0, stream>>>(pts, gidx, mean1, mean2, a1, be1, accums);
    k_buildx0<<<MROWS / 256, 256, 0, stream>>>(pts, gidx, mean1, mean2, new_pts, accums,
                                               a1, be1, a2, be2, xbuf);
    // layer 0: 128 -> 128
    k_cvtw<<<(128 * KIN) / 256, 256, 0, stream>>>(w[0], wf16, 128 * KIN);
    k_gemm<<<dim3(128 / 16, MROWS / 128), 256, 0, stream>>>(xbuf, wf16, bi[0], ybuf, 128);
    k_chanstats<<<512, 256, 0, stream>>>(ybuf, chan0, 128, (size_t)MROWS * 128);
    k_bnrelu<<<MROWS / 32, 256, 0, stream>>>(ybuf, chan0, gg[0], bbn[0], xbuf);
    // layer 1: 128 -> 128
    k_cvtw<<<(128 * KIN) / 256, 256, 0, stream>>>(w[1], wf16, 128 * KIN);
    k_gemm<<<dim3(128 / 16, MROWS / 128), 256, 0, stream>>>(xbuf, wf16, bi[1], ybuf, 128);
    k_chanstats<<<512, 256, 0, stream>>>(ybuf, chan1, 128, (size_t)MROWS * 128);
    k_bnrelu<<<MROWS / 32, 256, 0, stream>>>(ybuf, chan1, gg[1], bbn[1], xbuf);
    // layer 2: 128 -> 256, then BN + ReLU + max over nsample
    k_cvtw<<<(256 * KIN) / 256, 256, 0, stream>>>(w[2], wf16, 256 * KIN);
    k_gemm<<<dim3(256 / 16, MROWS / 128), 256, 0, stream>>>(xbuf, wf16, bi[2], ybuf, 256);
    k_chanstats<<<512, 256, 0, stream>>>(ybuf, chan2, 256, (size_t)MROWS * 256);
    k_final<<<BB * SS, 256, 0, stream>>>(ybuf, chan2, gg[2], bbn[2], out);
}
